// MultiheadA_17368847745519
// MI455X (gfx1250) — matrix operand order repaired and checked by reading, not compiled, not run on hardware
//
#include <hip/hip_runtime.h>
#include <hip/hip_bf16.h>
#include <math.h>

#define SEQ 2048
#define DIM 1024
#define NH  16
#define HD  64

typedef __attribute__((ext_vector_type(16))) _Float16 v16h;
typedef __attribute__((ext_vector_type(8)))  _Float16 v8h;
typedef __attribute__((ext_vector_type(8)))  float    v8f;
typedef __attribute__((ext_vector_type(4)))  int      v4i;

// gfx1250 async global->LDS path (same upstream series as s_wait_tensorcnt).
#if defined(__has_builtin)
#if __has_builtin(__builtin_amdgcn_global_load_async_to_lds_b128) && \
    __has_builtin(__builtin_amdgcn_s_wait_asynccnt)
#define USE_ASYNC_LDS 1
#endif
#endif
#ifndef USE_ASYNC_LDS
#define USE_ASYNC_LDS 0
#endif

static __device__ __forceinline__ v16h cat8(v8h lo, v8h hi) {
  v16h r;
#pragma unroll
  for (int i = 0; i < 8; ++i) { r[i] = lo[i]; r[i + 8] = hi[i]; }
  return r;
}

static __device__ __forceinline__ v8f wmma_f16(v16h a, v16h b, v8f c) {
  return __builtin_amdgcn_wmma_f32_16x16x32_f16(false, a, false, b, (short)0, c,
                                                false, false);
}

// 16-byte global -> LDS copy: async DMA when available, else load+ds_store.
static __device__ __forceinline__ void copy16(const _Float16* g, _Float16* l) {
#if USE_ASYNC_LDS
  __builtin_amdgcn_global_load_async_to_lds_b128((v4i*)(void*)g,
                                                 (v4i*)(void*)l, 0, 0);
#else
  *(v8h*)l = *(const v8h*)g;
#endif
}

static __device__ __forceinline__ void wait_async() {
#if USE_ASYNC_LDS
  __builtin_amdgcn_s_wait_asynccnt(0);
#endif
}

// Butterfly reductions over the 16 lanes of a half-wave (wave32).
static __device__ __forceinline__ float redmax16(float v) {
#pragma unroll
  for (int m = 1; m < 16; m <<= 1) v = fmaxf(v, __shfl_xor(v, m, 32));
  return v;
}
static __device__ __forceinline__ float redsum16(float v) {
#pragma unroll
  for (int m = 1; m < 16; m <<= 1) v += __shfl_xor(v, m, 32);
  return v;
}

// ---------------------------------------------------------------------------
// Phase 0: f32 -> f16 conversions + RoPE cos/sin tables (L2-resident staging)
// ---------------------------------------------------------------------------
__global__ void prep_convert(const float* __restrict__ x,
                             const float* __restrict__ Wq,
                             const float* __restrict__ Wk,
                             const float* __restrict__ Wv,
                             const float* __restrict__ Wo,
                             const float* __restrict__ inv_freq,
                             _Float16* __restrict__ xh,
                             _Float16* __restrict__ wqh,
                             _Float16* __restrict__ wkh,
                             _Float16* __restrict__ wvh,
                             _Float16* __restrict__ woh,
                             float* __restrict__ cosv,
                             float* __restrict__ sinv) {
  const long NX = (long)SEQ * DIM, NW = (long)DIM * DIM;
  const long TOT = NX + 4 * NW + (long)SEQ * 32;
  for (long i = (long)blockIdx.x * blockDim.x + threadIdx.x; i < TOT;
       i += (long)gridDim.x * blockDim.x) {
    if (i < NX) {
      xh[i] = (_Float16)x[i];
    } else if (i < NX + NW) {
      long j = i - NX; wqh[j] = (_Float16)Wq[j];
    } else if (i < NX + 2 * NW) {
      long j = i - NX - NW; wkh[j] = (_Float16)Wk[j];
    } else if (i < NX + 3 * NW) {
      long j = i - NX - 2 * NW; wvh[j] = (_Float16)Wv[j];
    } else if (i < NX + 4 * NW) {
      long j = i - NX - 3 * NW; woh[j] = (_Float16)Wo[j];
    } else {
      long j = i - NX - 4 * NW;
      int s = (int)(j >> 5), f = (int)(j & 31);
      float fr = (float)s * inv_freq[f];
      cosv[j] = cosf(fr);
      sinv[j] = sinf(fr);
    }
  }
}

// ---------------------------------------------------------------------------
// GEMM:  out[m,n] = sum_k A[m,k] * W[n,k]  (+ bias[n])  via WMMA f16->f32.
// One 16x16 tile per wave; block = 4 waves covering 64 columns.
// A/B fragments in contiguous k order (k = 16*(lane/16) + s), not ISA table order (wave32).
// ---------------------------------------------------------------------------
__global__ __launch_bounds__(128) void gemm_nt_wmma(
    const _Float16* __restrict__ A, const _Float16* __restrict__ W,
    const float* __restrict__ bias, float* __restrict__ out, int M, int N,
    int K) {
  const int wave = threadIdx.x >> 5, lane = threadIdx.x & 31;
  const int lg = lane >> 4, ln = lane & 15;
  const int m0 = blockIdx.x * 16;
  const int n0 = (blockIdx.y * 4 + wave) * 16;
  const _Float16* arow = A + (size_t)(m0 + ln) * K + lg * 16;
  const _Float16* wrow = W + (size_t)(n0 + ln) * K + lg * 16;
  v8f acc = {};
  for (int k0 = 0; k0 < K; k0 += 32) {
    __builtin_prefetch(arow + k0 + 256, 0, 0);  // -> global_prefetch_b8
    __builtin_prefetch(wrow + k0 + 256, 0, 0);
    v16h a = cat8(*(const v8h*)(arow + k0), *(const v8h*)(arow + k0 + 8));
    v16h b = *(const v16h*)(wrow + k0);
    acc = wmma_f16(a, b, acc);
  }
  float bn = bias ? bias[n0 + ln] : 0.0f;
#pragma unroll
  for (int r = 0; r < 8; ++r)
    out[(size_t)(m0 + r + lg * 8) * N + n0 + ln] = acc[r] + bn;
}

// ---------------------------------------------------------------------------
// RoPE + head split + scale fold + V transpose + zscale.
// ---------------------------------------------------------------------------
__global__ void rope_pack(const float* __restrict__ q_raw,
                          const float* __restrict__ k_raw,
                          const float* __restrict__ v_raw,
                          const float* __restrict__ cosv,
                          const float* __restrict__ sinv,
                          const float* __restrict__ Zf,
                          _Float16* __restrict__ qh, _Float16* __restrict__ kh,
                          _Float16* __restrict__ vT,
                          float* __restrict__ zscale) {
  int idx = blockIdx.x * blockDim.x + threadIdx.x;
  if (idx >= NH * SEQ * (HD / 2)) return;
  int d2 = idx & 31;
  int s = (idx >> 5) & (SEQ - 1);
  int h = idx >> 16;  // 32*2048 = 65536
  float c = cosv[s * 32 + d2], sn = sinv[s * 32 + d2];
  size_t base = (size_t)s * DIM + h * HD + 2 * d2;
  float qa = q_raw[base], qb = q_raw[base + 1];
  float ka = k_raw[base], kb = k_raw[base + 1];
  float va = v_raw[base], vb = v_raw[base + 1];
  const float sc = 0.35355339059327379f;  // 64^-0.25
  float qra = (qa * c - qb * sn), qrb = (qa * sn + qb * c);
  float kra = (ka * c - kb * sn), krb = (ka * sn + kb * c);
  size_t ob = ((size_t)h * SEQ + s) * HD + 2 * d2;
  qh[ob] = (_Float16)(qra * sc);
  qh[ob + 1] = (_Float16)(qrb * sc);
  kh[ob] = (_Float16)(kra * sc);
  kh[ob + 1] = (_Float16)(krb * sc);
  size_t vtb = ((size_t)h * HD + 2 * d2) * SEQ + s;
  vT[vtb] = (_Float16)va;
  vT[vtb + SEQ] = (_Float16)vb;
  if (d2 == 0) {
    float sp = log1pf(__expf(Zf[0]));     // softplus(Zfactor)
    sp = fminf(fmaxf(sp, 1e-5f), 1e-4f);  // clip
    zscale[h * SEQ + s] = (kra == 0.0f) ? sp : 1.0f;
  }
}

// ---------------------------------------------------------------------------
// Fused attention, flash-style. Block = 4 waves = 64 query rows of one head.
// K/V tiles for each 32-key chunk are staged cooperatively into LDS
// (double-buffered, async global->LDS DMA when available) and shared by all
// 4 waves -> 4x less L2 read traffic than per-wave fragment loads.
//   kbuf: 32 key rows x 64 halves, padded to 72 (16B-aligned, 16 bank groups)
//   vbuf: 64 hd rows  x 32 halves, padded to 40
//   pbuf: per-wave 16x32 probability tile (C/D layout -> A layout bounce)
// ---------------------------------------------------------------------------
__global__ __launch_bounds__(128) void attn_wmma(
    const _Float16* __restrict__ qh, const _Float16* __restrict__ kh,
    const _Float16* __restrict__ vT, const float* __restrict__ zscale,
    const float* __restrict__ mask, float* __restrict__ qk_out,
    _Float16* __restrict__ wv_h) {
  __shared__ _Float16 kbuf[2][32 * 72];
  __shared__ _Float16 vbuf[2][64 * 40];
  __shared__ _Float16 pbuf[4][16 * 40];
  const int tid = threadIdx.x;
  const int wave = tid >> 5, lane = tid & 31;
  const int lg = lane >> 4, ln = lane & 15;
  const int h = blockIdx.y;
  const int m0 = blockIdx.x * 64 + wave * 16;

  const _Float16* khp = kh + (size_t)h * SEQ * HD;  // [S][hd]
  const _Float16* vtp = vT + (size_t)h * HD * SEQ;  // [hd][S]

  // Cooperative stage of 32-key chunk kt into buffer b (8 KB, 64 B/thread).
  auto stage = [&](int kt, int b) {
#pragma unroll
    for (int i = 0; i < 2; ++i) {
      const int u = tid + i * 128;
      {  // K tile: 32 rows x 8 units of 16 B
        const int row = u >> 3, col = u & 7;
        copy16(khp + (size_t)(kt * 32 + row) * HD + col * 8,
               &kbuf[b][row * 72 + col * 8]);
      }
      {  // V tile: 64 rows x 4 units of 16 B
        const int row = u >> 2, col = u & 3;
        copy16(vtp + (size_t)row * SEQ + kt * 32 + col * 8,
               &vbuf[b][row * 40 + col * 8]);
      }
    }
  };

  const _Float16* qrow = qh + ((size_t)h * SEQ + m0 + ln) * HD + lg * 16;
  v16h qa0 = cat8(*(const v8h*)(qrow), *(const v8h*)(qrow + 8));
  v16h qa1 = cat8(*(const v8h*)(qrow + 32), *(const v8h*)(qrow + 40));

  float mx[8], lsum[8];
#pragma unroll
  for (int r = 0; r < 8; ++r) { mx[r] = -3.0e38f; lsum[r] = 0.0f; }
  v8f o[4] = {};

  stage(0, 0);
  for (int kt = 0; kt < SEQ / 32; ++kt) {
    const int cur = kt & 1;
    wait_async();
    __syncthreads();  // buffer[cur] staged & visible block-wide
    if (kt + 1 < SEQ / 32) stage(kt + 1, cur ^ 1);

    float pv[2][8];
#pragma unroll
    for (int sub = 0; sub < 2; ++sub) {
      const int n0 = kt * 32 + sub * 16;
      const _Float16* krow = &kbuf[cur][(sub * 16 + ln) * 72 + lg * 16];
      v16h kb0 = *(const v16h*)(krow);
      v16h kb1 = *(const v16h*)(krow + 32);
      v8f acc = {};
      acc = wmma_f16(qa0, kb0, acc);
      acc = wmma_f16(qa1, kb1, acc);
      const float zs = zscale[h * SEQ + n0 + ln];
      const size_t col = n0 + ln;
#pragma unroll
      for (int r = 0; r < 8; ++r) {
        const int m = m0 + r + lg * 8;
        float val = (acc[r] + mask[(size_t)m * SEQ + col] * zs) * zs;
        qk_out[((size_t)h * SEQ + m) * SEQ + col] = val;
        pv[sub][r] = val;
      }
    }
    // online softmax update + stage probabilities to per-wave LDS tile
    float corr[8];
#pragma unroll
    for (int r = 0; r < 8; ++r) {
      float tm = redmax16(fmaxf(pv[0][r], pv[1][r]));
      float nm = fmaxf(mx[r], tm);
      float c = __expf(mx[r] - nm);
      float p0 = __expf(pv[0][r] - nm);
      float p1 = __expf(pv[1][r] - nm);
      lsum[r] = lsum[r] * c + redsum16(p0 + p1);
      mx[r] = nm;
      corr[r] = c;
      pbuf[wave][(r + lg * 8) * 40 + ln] = (_Float16)p0;
      pbuf[wave][(r + lg * 8) * 40 + 16 + ln] = (_Float16)p1;
    }
#pragma unroll
    for (int nt = 0; nt < 4; ++nt)
#pragma unroll
      for (int r = 0; r < 8; ++r) o[nt][r] *= corr[r];
    __syncthreads();  // p-tile write -> read fence
    v16h pa = cat8(*(const v8h*)(&pbuf[wave][ln * 40 + lg * 16]),
                   *(const v8h*)(&pbuf[wave][ln * 40 + lg * 16 + 8]));
#pragma unroll
    for (int nt = 0; nt < 4; ++nt) {
      const _Float16* vrow = &vbuf[cur][(nt * 16 + ln) * 40 + lg * 16];
      v16h vb = *(const v16h*)(vrow);
      o[nt] = wmma_f16(pa, vb, o[nt]);
    }
    __syncthreads();  // all reads of buffer[cur]/pbuf done before overwrite
  }
#pragma unroll
  for (int r = 0; r < 8; ++r) {
    const float inv = 1.0f / lsum[r];
#pragma unroll
    for (int nt = 0; nt < 4; ++nt)
      wv_h[(size_t)(m0 + r + lg * 8) * DIM + h * HD + nt * 16 + ln] =
          (_Float16)(o[nt][r] * inv);
  }
}

// ---------------------------------------------------------------------------
extern "C" void kernel_launch(void* const* d_in, const int* in_sizes, int n_in,
                              void* d_out, int out_size, void* d_ws,
                              size_t ws_size, hipStream_t stream) {
  const float* x = (const float*)d_in[0];
  const float* mask = (const float*)d_in[1];
  const float* Wq = (const float*)d_in[2];
  const float* bq = (const float*)d_in[3];
  const float* Wk = (const float*)d_in[4];
  const float* Wv = (const float*)d_in[5];
  const float* bv = (const float*)d_in[6];
  const float* Wo = (const float*)d_in[7];
  const float* bo = (const float*)d_in[8];
  const float* Zf = (const float*)d_in[9];
  const float* inv_freq = (const float*)d_in[10];

  float* out = (float*)d_out;           // [S][D]
  float* qk = out + (size_t)SEQ * DIM;  // [H][S][S]

  char* p = (char*)d_ws;
  auto carve = [&](size_t bytes) -> char* {
    char* r = p;
    p += (bytes + 255) & ~(size_t)255;
    return r;
  };
  _Float16* xh = (_Float16*)carve((size_t)SEQ * DIM * 2);
  _Float16* wqh = (_Float16*)carve((size_t)DIM * DIM * 2);
  _Float16* wkh = (_Float16*)carve((size_t)DIM * DIM * 2);
  _Float16* wvh = (_Float16*)carve((size_t)DIM * DIM * 2);
  _Float16* woh = (_Float16*)carve((size_t)DIM * DIM * 2);
  float* cosv = (float*)carve((size_t)SEQ * 32 * 4);
  float* sinv = (float*)carve((size_t)SEQ * 32 * 4);
  float* q_raw = (float*)carve((size_t)SEQ * DIM * 4);
  float* k_raw = (float*)carve((size_t)SEQ * DIM * 4);
  float* v_raw = (float*)carve((size_t)SEQ * DIM * 4);
  _Float16* qh = (_Float16*)carve((size_t)NH * SEQ * HD * 2);
  _Float16* kh = (_Float16*)carve((size_t)NH * SEQ * HD * 2);
  _Float16* vT = (_Float16*)carve((size_t)NH * HD * SEQ * 2);
  float* zscale = (float*)carve((size_t)NH * SEQ * 4);
  _Float16* wv_h = (_Float16*)carve((size_t)SEQ * DIM * 2);

  prep_convert<<<8192, 256, 0, stream>>>(x, Wq, Wk, Wv, Wo, inv_freq, xh, wqh,
                                         wkh, wvh, woh, cosv, sinv);

  dim3 gg(SEQ / 16, DIM / 64);
  gemm_nt_wmma<<<gg, 128, 0, stream>>>(xh, wqh, bq, q_raw, SEQ, DIM, DIM);
  gemm_nt_wmma<<<gg, 128, 0, stream>>>(xh, wkh, nullptr, k_raw, SEQ, DIM, DIM);
  gemm_nt_wmma<<<gg, 128, 0, stream>>>(xh, wvh, bv, v_raw, SEQ, DIM, DIM);

  rope_pack<<<(NH * SEQ * 32) / 256, 256, 0, stream>>>(
      q_raw, k_raw, v_raw, cosv, sinv, Zf, qh, kh, vT, zscale);

  attn_wmma<<<dim3(SEQ / 64, NH), 128, 0, stream>>>(qh, kh, vT, zscale, mask,
                                                    qk, wv_h);

  gemm_nt_wmma<<<gg, 128, 0, stream>>>(wv_h, woh, bo, out, SEQ, DIM, DIM);
}
